// GNNModel_7490422964320
// MI455X (gfx1250) — matrix operand order repaired and checked by reading, not compiled, not run on hardware
//
#include <hip/hip_runtime.h>
#include <hip/hip_bf16.h>
#include <stdint.h>

#define D 128   // feature width (D_IN == D_H == 128)

typedef __attribute__((ext_vector_type(16))) __bf16 v16bf;
typedef __attribute__((ext_vector_type(8)))  float  v8f;

// POD 16-byte vector so it can live inside a union (HIP's uint4 has ctors).
struct __attribute__((aligned(16))) u32x4 { unsigned int x, y, z, w; };

static __device__ __forceinline__ unsigned short f32_to_bf16_rne(float f) {
    unsigned int u = __float_as_uint(f);
    unsigned int r = 0x7FFFu + ((u >> 16) & 1u);
    return (unsigned short)((u + r) >> 16);
}

// ---------------- degree / norm ----------------
__global__ void k_deg_init(float* __restrict__ deg, int n) {
    int i = blockIdx.x * blockDim.x + threadIdx.x;
    if (i < n) deg[i] = 1.0f;                    // self-loop counts once
}

__global__ void k_deg_count(const int* __restrict__ ei, float* __restrict__ deg, int E) {
    int e = blockIdx.x * blockDim.x + threadIdx.x;
    if (e < E) atomicAdd(&deg[ei[E + e]], 1.0f); // dst = ei[1][e]
}

__global__ void k_rsqrt_inplace(float* __restrict__ deg, int n) {
    int i = blockIdx.x * blockDim.x + threadIdx.x;
    if (i < n) deg[i] = rsqrtf(deg[i]);          // deg >= 1 always
}

// ---------------- dtype helpers ----------------
__global__ void k_f32_to_bf16(const float* __restrict__ x, unsigned short* __restrict__ xb, int n) {
    int i = blockIdx.x * blockDim.x + threadIdx.x;
    if (i < n) xb[i] = f32_to_bf16_rne(x[i]);
}

__global__ void k_zero_f32(float* __restrict__ p, int n) {
    int i = blockIdx.x * blockDim.x + threadIdx.x;
    if (i < n) p[i] = 0.0f;
}

// Pack W [128(k) x 128(n)] f32 -> bf16 B-fragments.
// Element index: ((ntile*4 + chunk)*32 + lane)*16 + e
// lane = n%16 + 16*half ; element e holds B[chunk*32 + half*16 + e, n]  (contiguous k order, same as the A fragment)
__global__ void k_pack_w(const float* __restrict__ W, unsigned short* __restrict__ Wp) {
    int idx = blockIdx.x * blockDim.x + threadIdx.x;   // 16384 elements
    int e     =  idx        & 15;
    int lane  = (idx >> 4)  & 31;
    int chunk = (idx >> 9)  & 3;
    int ntile =  idx >> 11;
    int n = (ntile << 4) + (lane & 15);
    int k = (chunk << 5) + ((lane >> 4) << 4) + e;
    Wp[idx] = f32_to_bf16_rne(W[k * D + n]);
}

// ---------------- WMMA GEMM: H[M,128] = A[M,128] @ W[128,128] ----------------
// Block = 4 waves; packed W staged once into LDS (32 KB of the 320 KB/WGP).
// Each wave computes one 16x128 output strip: 8 n-tiles x 4 k-chunks = 32 WMMAs.
// Per k-chunk, all 8 B fragments are loaded into distinct registers first (LDS
// returns in order -> incremental s_wait_dscnt), then 8 WMMAs issue back-to-back.
__global__ void __launch_bounds__(128) k_gemm_wmma(
    const unsigned short* __restrict__ Abf,   // [M,128] bf16 row-major
    const unsigned short* __restrict__ Wp,    // packed fragments (16384 bf16)
    float* __restrict__ H, int M)
{
    __shared__ unsigned short sW[8 * 4 * 32 * 16];   // 32 KB

    const int lane = threadIdx.x;            // 0..31
    const int wv   = threadIdx.y;            // 0..3
    const int tid  = wv * 32 + lane;         // 0..127

    // cooperative stage: 2048 x 16B
    {
        const u32x4* src = (const u32x4*)Wp;
        u32x4*       dst = (u32x4*)sW;
#pragma unroll
        for (int i = 0; i < 16; ++i)
            dst[tid + i * 128] = src[tid + i * 128];
    }
    __syncthreads();

    const int m0 = (blockIdx.x * 4 + wv) * 16;
    if (m0 >= M) return;                     // wave-uniform; EXEC all-ones inside

    const int half = lane >> 4;
    const unsigned short* arow = Abf + (size_t)(m0 + (lane & 15)) * D;
    const unsigned short* brow = sW + (lane << 4);   // per-lane base into fragments

    union Frag { v16bf v; u32x4 q[2]; };
    Frag a, b[8];
    v8f acc[8];
#pragma unroll
    for (int nt = 0; nt < 8; ++nt) acc[nt] = (v8f){0.f,0.f,0.f,0.f,0.f,0.f,0.f,0.f};

#pragma unroll
    for (int c = 0; c < 4; ++c) {
        const int kb = c << 5;
        // A 16x32 bf16 fragment: e0..15 = K kb+half*16..kb+half*16+15
        a.q[0] = *(const u32x4*)(arow + kb + half * 16);
        a.q[1] = *(const u32x4*)(arow + kb + half * 16 + 8);

        // load all 8 B fragments of this k-chunk into distinct registers
#pragma unroll
        for (int nt = 0; nt < 8; ++nt) {
            const unsigned short* wp = brow + (((nt << 2) + c) << 9);
            b[nt].q[0] = *(const u32x4*)(wp);
            b[nt].q[1] = *(const u32x4*)(wp + 8);
        }
        // 8 independent WMMAs (distinct acc + distinct B regs)
#pragma unroll
        for (int nt = 0; nt < 8; ++nt)
            acc[nt] = __builtin_amdgcn_wmma_f32_16x16x32_bf16(
                          false, a.v, false, b[nt].v, (short)0, acc[nt], false, false);
    }

    // D layout: VGPR r -> row m0 + half*8 + r, col ntile*16 + lane%16
    float* outbase = H + (size_t)(m0 + half * 8) * D + (lane & 15);
    if (m0 + 16 <= M) {                      // uniform fast path (always true: M % 16 == 0)
#pragma unroll
        for (int nt = 0; nt < 8; ++nt)
#pragma unroll
            for (int r = 0; r < 8; ++r)
                outbase[(size_t)r * D + (nt << 4)] = acc[nt][r];
    } else {                                 // safety tail, never hit for M = 50000
        for (int nt = 0; nt < 8; ++nt)
            for (int r = 0; r < 8; ++r)
                if (m0 + half * 8 + r < M)
                    outbase[(size_t)r * D + (nt << 4)] = acc[nt][r];
    }
}

// ---------------- edge scatter: Acc[dst] += H[src] * dinv[src]*dinv[dst] ----------------
// One wave per edge; each lane handles 4 floats (32*4 = 128). Working set (H + Acc
// = 51 MB) sits in the 192 MB L2, so gathers and f32 atomics are L2-resident.
__global__ void __launch_bounds__(256) k_scatter(
    const int* __restrict__ ei, const float* __restrict__ dinv,
    const float* __restrict__ H, float* __restrict__ Acc, int E)
{
    int e = blockIdx.x * 8 + threadIdx.y;
    if (e >= E) return;
    int s = ei[e];
    int d = ei[E + e];
    float nrm = dinv[s] * dinv[d];
    int lane = threadIdx.x;
    float4 h4 = ((const float4*)(H + (size_t)s * D))[lane];
    float* dst = Acc + (size_t)d * D + (lane << 2);
    atomicAdd(dst + 0, h4.x * nrm);
    atomicAdd(dst + 1, h4.y * nrm);
    atomicAdd(dst + 2, h4.z * nrm);
    atomicAdd(dst + 3, h4.w * nrm);
}

// ---------------- finalize: self-loop + bias + ReLU ----------------
__global__ void k_finalize_mid(const float* __restrict__ Acc, const float* __restrict__ H,
                               const float* __restrict__ dinv, const float* __restrict__ bias,
                               unsigned short* __restrict__ Xb, int total)
{
    int idx = blockIdx.x * blockDim.x + threadIdx.x;
    if (idx >= total) return;
    int i = idx >> 7, dd = idx & (D - 1);
    float di = dinv[i];
    float v = Acc[idx] + H[idx] * di * di + bias[dd];
    v = fmaxf(v, 0.0f);
    Xb[idx] = f32_to_bf16_rne(v);            // next layer's GEMM input
}

__global__ void k_finalize_last(const float* __restrict__ Acc, const float* __restrict__ H,
                                const float* __restrict__ dinv, const float* __restrict__ bias,
                                const int* __restrict__ batch,
                                float* __restrict__ node_out, float* __restrict__ graph_out,
                                int total)
{
    int idx = blockIdx.x * blockDim.x + threadIdx.x;
    if (idx >= total) return;
    int i = idx >> 7, dd = idx & (D - 1);
    float di = dinv[i];
    float v = Acc[idx] + H[idx] * di * di + bias[dd];
    v = fmaxf(v, 0.0f);
    node_out[idx] = v;
    atomicAdd(&graph_out[(size_t)batch[i] * D + dd], v);   // sum readout
}

// ---------------- driver ----------------
extern "C" void kernel_launch(void* const* d_in, const int* in_sizes, int n_in,
                              void* d_out, int out_size, void* d_ws, size_t ws_size,
                              hipStream_t stream)
{
    const float* x     = (const float*)d_in[0];
    const int*   ei    = (const int*)  d_in[1];
    const int*   batch = (const int*)  d_in[2];
    const float* W[3]  = {(const float*)d_in[3], (const float*)d_in[5], (const float*)d_in[7]};
    const float* B[3]  = {(const float*)d_in[4], (const float*)d_in[6], (const float*)d_in[8]};

    const int N     = in_sizes[0] / D;      // 50000
    const int E     = in_sizes[1] / 2;      // 800000
    const int total = N * D;
    float* node_out  = (float*)d_out;
    float* graph_out = node_out + (size_t)total;
    const int gfeat  = out_size - total;    // G*D = 4096

    // workspace carve-up (all 256B aligned): ~64.3 MB
    char* ws = (char*)d_ws;
    auto al = [](size_t v) { return (v + 255) & ~(size_t)255; };
    float*          dinv = (float*)ws;          ws += al((size_t)N * sizeof(float));
    unsigned short* xb   = (unsigned short*)ws; ws += al((size_t)total * sizeof(unsigned short));
    float*          h    = (float*)ws;          ws += al((size_t)total * sizeof(float));
    float*          acc  = (float*)ws;          ws += al((size_t)total * sizeof(float));
    unsigned short* wp   = (unsigned short*)ws; ws += al((size_t)(8 * 4 * 32 * 16) * sizeof(unsigned short));

    const dim3 b256(256);
    const dim3 sblk(32, 8);                 // scatter: 8 waves / block, 1 wave / edge
    const dim3 gblk(32, 4);                 // gemm: 4 waves / block, 1 strip / wave
    const int  mtiles = (N + 15) / 16;      // 3125

    // structure-only normalization (reused per layer)
    k_deg_init     <<<(N + 255) / 256, b256, 0, stream>>>(dinv, N);
    k_deg_count    <<<(E + 255) / 256, b256, 0, stream>>>(ei, dinv, E);
    k_rsqrt_inplace<<<(N + 255) / 256, b256, 0, stream>>>(dinv, N);
    k_f32_to_bf16  <<<(total + 255) / 256, b256, 0, stream>>>(x, xb, total);

    for (int l = 0; l < 3; ++l) {
        k_pack_w   <<<64, b256, 0, stream>>>(W[l], wp);
        k_gemm_wmma<<<(mtiles + 3) / 4, gblk, 0, stream>>>(xb, wp, h, N);
        k_zero_f32 <<<(total + 255) / 256, b256, 0, stream>>>(acc, total);
        k_scatter  <<<(E + 7) / 8, sblk, 0, stream>>>(ei, dinv, h, acc, E);
        if (l < 2) {
            k_finalize_mid<<<(total + 255) / 256, b256, 0, stream>>>(acc, h, dinv, B[l], xb, total);
        } else {
            k_zero_f32     <<<(gfeat + 255) / 256, b256, 0, stream>>>(graph_out, gfeat);
            k_finalize_last<<<(total + 255) / 256, b256, 0, stream>>>(acc, h, dinv, B[l], batch,
                                                                      node_out, graph_out, total);
        }
    }
    (void)n_in; (void)ws_size;
}
